// SearchTransfer_61306363183487
// MI455X (gfx1250) — matrix operand order repaired and checked by reading, not compiled, not run on hardware
//
#include <hip/hip_runtime.h>
#include <hip/hip_bf16.h>

typedef _Float16 v16h __attribute__((ext_vector_type(16)));
typedef _Float16 h8   __attribute__((ext_vector_type(8)));
typedef float    v8f  __attribute__((ext_vector_type(8)));

#define BM 128
#define BN 128
#define BK 32

// ---------------------------------------------------------------------------
// Swizzled B layout: fragment-ready order for v_wmma_f32_16x16x32_f16.
// For logical B[k][n] (K x N): fragment tile (kt = k/32, nt = n/16),
// lane = (k%32 >= 16 ? 16 : 0) + n%16, half h = k%16.
// Linear index = ((kt * (N/16) + nt) * 32 + lane) * 16 + h.
// A wave loads a whole 32x16 B fragment as 32 contiguous bytes per lane.
// ---------------------------------------------------------------------------
__device__ __forceinline__ size_t bswz_index(int k, int n, int N) {
    int kt = k >> 5, kr = k & 31;
    int nt = n >> 4, nr = n & 15;
    int lane = (kr & 16) + nr;
    int h = kr & 15;
    return ((((size_t)kt * (size_t)(N >> 4) + nt) * 32) + (size_t)lane) * 16 + h;
}

// 16-byte per-lane async copy global -> LDS (CDNA5, tracked by ASYNCcnt)
__device__ __forceinline__ void async_copy_b128(uint32_t lds_off, const void* gptr) {
    asm volatile("global_load_async_to_lds_b128 %0, %1, off"
                 :: "v"(lds_off), "v"(gptr) : "memory");
}

// ---------------------------------------------------------------------------
// f16 WMMA GEMM:  C[M][N] = A[M][K] * Bsw (+ bias[N])
// A row-major f16 (async-staged through LDS, double buffered),
// Bsw in swizzled fragment layout (read straight from global / L2).
// Requires M%128==0, N%128==0, K%32==0 (true for all call sites here).
// ---------------------------------------------------------------------------
__global__ __launch_bounds__(256) void gemm_f16_wmma(
    const _Float16* __restrict__ A,
    const _Float16* __restrict__ Bsw,
    const float*    __restrict__ bias,
    float*          __restrict__ Cf,
    _Float16*       __restrict__ Ch,
    int M, int N, int K)
{
    __shared__ __align__(16) _Float16 As[2][BM][BK];

    const int tid  = threadIdx.x;
    const int lane = tid & 31;
    const int wave = tid >> 5;        // 0..7
    const int wrow = wave >> 2;       // 0..1 : 64 rows of C each
    const int wcol = wave & 3;        // 0..3 : 32 cols of C each

    const int bm = blockIdx.y * BM;
    const int bn = blockIdx.x * BN;

    const int  mrow   = lane & 15;
    const bool hiHalf = lane >= 16;
    const int  kb0    = hiHalf ? 16 : 0;   // A fragment contiguous K bases
    const int  kb1    = hiHalf ? 24 : 8;
    const int  ntiles = N >> 4;
    const int  nk     = K / BK;

    // Per-thread async chunk mapping: 512 chunks of 16B per A tile,
    // 2 chunks per thread (one per asm issue; 2 instructions per wave).
    const int ar0 = tid >> 2,          ac0 = (tid & 3) << 3;
    const int ar1 = (tid + 256) >> 2,  ac1 = ((tid + 256) & 3) << 3;
    const _Float16* ag0 = A + (size_t)(bm + ar0) * K + ac0;
    const _Float16* ag1 = A + (size_t)(bm + ar1) * K + ac1;
    const uint32_t lbase = (uint32_t)(uintptr_t)(&As[0][0][0]);
    const uint32_t l0 = lbase + (uint32_t)(ar0 * (BK * 2) + ac0 * 2);
    const uint32_t l1 = lbase + (uint32_t)(ar1 * (BK * 2) + ac1 * 2);
    const uint32_t bufBytes = BM * BK * 2;

    v8f acc[4][2];
#pragma unroll
    for (int i = 0; i < 4; ++i)
#pragma unroll
        for (int j = 0; j < 2; ++j)
#pragma unroll
            for (int e = 0; e < 8; ++e) acc[i][j][e] = 0.0f;

    // prologue: A tile 0 -> buffer 0
    async_copy_b128(l0, (const void*)ag0);
    async_copy_b128(l1, (const void*)ag1);

    for (int kt = 0; kt < nk; ++kt) {
        const int buf = kt & 1;

        // stream next A tile into the other buffer, then counted wait so the
        // current tile is complete while the next stays in flight.
        if (kt + 1 < nk) {
            uint32_t off = (uint32_t)(buf ^ 1) * bufBytes;
            async_copy_b128(l0 + off, (const void*)(ag0 + (size_t)(kt + 1) * BK));
            async_copy_b128(l1 + off, (const void*)(ag1 + (size_t)(kt + 1) * BK));
            asm volatile("s_wait_asynccnt 2" ::: "memory");
        } else {
            asm volatile("s_wait_asynccnt 0" ::: "memory");
        }
        __syncthreads();   // all waves' current A tile resident

        // --- A fragments: two contiguous b128 LDS loads per 16x32 tile ---
        union { v16h v; h8 p[2]; } af[4];
#pragma unroll
        for (int mt = 0; mt < 4; ++mt) {
            int m = wrow * 64 + mt * 16 + mrow;
            af[mt].p[0] = *(const h8*)&As[buf][m][kb0];
            af[mt].p[1] = *(const h8*)&As[buf][m][kb1];
        }
        // --- B fragments: two contiguous b128 global loads (L2 resident) --
        union { v16h v; h8 p[2]; } bf[2];
#pragma unroll
        for (int nt = 0; nt < 2; ++nt) {
            int tn = (bn >> 4) + wcol * 2 + nt;
            const _Float16* bp =
                Bsw + ((((size_t)kt * ntiles + tn) * 32) + lane) * 16;
            bf[nt].p[0] = *(const h8*)bp;
            bf[nt].p[1] = *(const h8*)(bp + 8);
        }

        // --- 8 WMMAs per wave per k-step ---------------------------------
#pragma unroll
        for (int mt = 0; mt < 4; ++mt)
#pragma unroll
            for (int nt = 0; nt < 2; ++nt)
                acc[mt][nt] = __builtin_amdgcn_wmma_f32_16x16x32_f16(
                    false, af[mt].v, false, bf[nt].v,
                    (short)0, acc[mt][nt], false, false);

        __syncthreads();   // buffer reads done before it is overwritten
    }

    // --- epilogue: bias add, store f32 or f16 ----------------------------
#pragma unroll
    for (int mt = 0; mt < 4; ++mt)
#pragma unroll
        for (int nt = 0; nt < 2; ++nt) {
            int col = bn + wcol * 32 + nt * 16 + mrow;
            float bv = bias ? bias[col] : 0.0f;
#pragma unroll
            for (int i = 0; i < 8; ++i) {
                int row = bm + wrow * 64 + mt * 16 + i + (hiHalf ? 8 : 0);
                float v = acc[mt][nt][i] + bv;
                if (Cf) Cf[(size_t)row * N + col] = v;
                else    Ch[(size_t)row * N + col] = (_Float16)v;
            }
        }
}

// ---------------------------------------------------------------------------
// 3x3 / pad 1 / stride 1 unfold of a [C][H][H] f32 tensor.
// One block per patch n (grid = H*H), 256 threads over Kdim = C*9.
// mode 0: raw f16, A layout dst[n*Kdim + k]
// mode 1: raw f16 A layout + invnorm[n] = 1/max(||col||,eps)
// mode 2: normalized f16, swizzled-B layout (patch index = column n)
// ---------------------------------------------------------------------------
__global__ __launch_bounds__(256) void unfold3_kernel(
    const float* __restrict__ src, int C, int H,
    _Float16* __restrict__ dst, float* __restrict__ invnorm, int mode)
{
    const int n = blockIdx.x;
    const int Np = H * H;
    const int y = n / H, x = n % H;
    const int Kdim = C * 9;

    float vals[12];
    int   kidx[12];
    int   cnt = 0;
    float ss = 0.0f;
    for (int k = threadIdx.x; k < Kdim; k += 256) {
        int c = k / 9;
        int r = k - c * 9;
        int ky = r / 3, kx = r - ky * 3;
        int sy = y + ky - 1, sx = x + kx - 1;
        float v = 0.0f;
        if (sy >= 0 && sy < H && sx >= 0 && sx < H)
            v = src[((size_t)c * H + sy) * H + sx];
        vals[cnt] = v; kidx[cnt] = k; ++cnt;
        ss += v * v;
    }

    __shared__ float red[256];
    red[threadIdx.x] = ss;
    __syncthreads();
    for (int s = 128; s > 0; s >>= 1) {
        if (threadIdx.x < s) red[threadIdx.x] += red[threadIdx.x + s];
        __syncthreads();
    }
    float inv = 1.0f / fmaxf(sqrtf(red[0]), 1e-12f);
    if (mode == 1 && threadIdx.x == 0) invnorm[n] = inv;
    float scale = (mode == 2) ? inv : 1.0f;

    for (int i = 0; i < cnt; ++i) {
        int k = kidx[i];
        if (mode == 2) dst[bswz_index(k, n, Np)]   = (_Float16)(vals[i] * scale);
        else           dst[(size_t)n * Kdim + k]   = (_Float16)vals[i];
    }
}

// W: [N][K] f32 row-major  ->  swizzled-B f16 (transposed convert)
__global__ __launch_bounds__(256) void pack_w_swizzled(
    const float* __restrict__ W, _Float16* __restrict__ Bt, int N, int K)
{
    int i = blockIdx.x * 256 + threadIdx.x;
    if (i >= N * K) return;
    int k = i / N, n = i - k * N;
    Bt[bswz_index(k, n, N)] = (_Float16)W[(size_t)n * K + k];
}

// invnorm[n] = 1/max(||row n of A||, eps), A f16 [Nrows][Kdim]
__global__ __launch_bounds__(256) void row_invnorm_f16(
    const _Float16* __restrict__ A, int Kdim, float* __restrict__ invnorm)
{
    int n = blockIdx.x;
    const _Float16* row = A + (size_t)n * Kdim;
    float ss = 0.0f;
    for (int k = threadIdx.x; k < Kdim; k += 256) {
        float v = (float)row[k];
        ss += v * v;
    }
    __shared__ float red[256];
    red[threadIdx.x] = ss;
    __syncthreads();
    for (int s = 128; s > 0; s >>= 1) {
        if (threadIdx.x < s) red[threadIdx.x] += red[threadIdx.x + s];
        __syncthreads();
    }
    if (threadIdx.x == 0) invnorm[n] = 1.0f / fmaxf(sqrtf(red[0]), 1e-12f);
}

// Per-column max (and optional argmax) of S[Nrows][Ncols], each row scaled
// by invnorm[n]. One block per column.
__global__ __launch_bounds__(256) void colmax_kernel(
    const float* __restrict__ S, const float* __restrict__ invnorm,
    int Nrows, int Ncols, float* __restrict__ best, int* __restrict__ bestArg)
{
    int m = blockIdx.x;
    float bv = -3.4e38f;
    int   ba = 0;
    for (int n = threadIdx.x; n < Nrows; n += 256) {
        float v = S[(size_t)n * Ncols + m] * invnorm[n];
        if (v > bv) { bv = v; ba = n; }
    }
    __shared__ float rv[256];
    __shared__ int   ra[256];
    rv[threadIdx.x] = bv; ra[threadIdx.x] = ba;
    __syncthreads();
    for (int s = 128; s > 0; s >>= 1) {
        if (threadIdx.x < s) {
            if (rv[threadIdx.x + s] > rv[threadIdx.x]) {
                rv[threadIdx.x] = rv[threadIdx.x + s];
                ra[threadIdx.x] = ra[threadIdx.x + s];
            }
        }
        __syncthreads();
    }
    if (threadIdx.x == 0) {
        best[m] = rv[0];
        if (bestArg) bestArg[m] = ra[0];
    }
}

// Gather best-matching ref patches by arg[] and overlap-add fold, /9.
// Patch grid is always 32x32; ref is [C][H][H] with H == Hout.
__global__ __launch_bounds__(256) void gather_fold_kernel(
    const float* __restrict__ ref, const int* __restrict__ arg,
    int C, int Hout, int ksz, int pad, int stride,
    float* __restrict__ out)
{
    int total = C * Hout * Hout;
    int i = blockIdx.x * 256 + threadIdx.x;
    if (i >= total) return;
    int x = i % Hout;
    int y = (i / Hout) % Hout;
    int c = i / (Hout * Hout);

    float sum = 0.0f;
    for (int ky = 0; ky < ksz; ++ky) {
        int ty = y + pad - ky;
        if (ty < 0 || (ty % stride) != 0) continue;
        int py = ty / stride;
        if (py >= 32) continue;
        for (int kx = 0; kx < ksz; ++kx) {
            int tx = x + pad - kx;
            if (tx < 0 || (tx % stride) != 0) continue;
            int px = tx / stride;
            if (px >= 32) continue;
            int n  = arg[py * 32 + px];
            int ny = n >> 5, nx = n & 31;
            int sy = ny * stride + ky - pad;
            int sx = nx * stride + kx - pad;
            if (sy >= 0 && sy < Hout && sx >= 0 && sx < Hout)
                sum += ref[((size_t)c * Hout + sy) * Hout + sx];
        }
    }
    out[i] = sum * (1.0f / 9.0f);
}

// ---------------------------------------------------------------------------
extern "C" void kernel_launch(void* const* d_in, const int* in_sizes, int n_in,
                              void* d_out, int out_size, void* d_ws, size_t ws_size,
                              hipStream_t stream)
{
    const float* lrsr_lv3  = (const float*)d_in[0];
    const float* refsr_lv1 = (const float*)d_in[1];
    const float* refsr_lv2 = (const float*)d_in[2];
    const float* refsr_lv3 = (const float*)d_in[3];
    const float* ref_lv1   = (const float*)d_in[4];
    const float* ref_lv2   = (const float*)d_in[5];
    const float* ref_lv3   = (const float*)d_in[6];
    const float* W1        = (const float*)d_in[7];
    const float* b1        = (const float*)d_in[8];
    const float* W2        = (const float*)d_in[9];
    const float* b2        = (const float*)d_in[10];
    float* out = (float*)d_out;

    // ---- workspace layout (256B aligned regions) ------------------------
    char* ws = (char*)d_ws;
    size_t off = 0;
    auto alloc = [&](size_t bytes) -> void* {
        void* p = ws + off;
        off = (off + bytes + 255) & ~(size_t)255;
        return p;
    };

    // Scores buffer (max 16384x1024 f32). Staging buffers that are dead
    // before any score GEMM runs are aliased inside it.
    float*     scores = (float*)alloc((size_t)16384 * 1024 * 4);   // 67.1 MB
    _Float16*  u2  = (_Float16*)scores;                             // 4096x1152
    _Float16*  B1t = (_Float16*)((char*)scores + 9437184);          // 1152x2304 swz
    _Float16*  u1  = (_Float16*)((char*)scores + 14745600);         // 16384x576
    _Float16*  B2t = (_Float16*)((char*)scores + 33619968);         // 576x2304 swz

    _Float16* lr3n = (_Float16*)alloc((size_t)2304 * 1024 * 2);     // swz B, normed
    _Float16* A3   = (_Float16*)alloc((size_t)1024 * 2304 * 2);     // raw r3 rows
    _Float16* r2   = (_Float16*)alloc((size_t)4096 * 2304 * 2);
    _Float16* r1   = (_Float16*)alloc((size_t)16384 * 2304 * 2);
    float* inv3 = (float*)alloc(1024 * 4);
    float* inv2 = (float*)alloc(4096 * 4);
    float* inv1 = (float*)alloc(16384 * 4);
    int*   arg  = (int*)alloc(1024 * 4);

    // ---- output offsets (floats): S_1, S_2, S_3, T_lv3, T_lv2, T_lv1 ----
    float* S1o = out + 0;
    float* S2o = out + 1024;
    float* S3o = out + 2048;
    float* T3o = out + 3072;
    float* T2o = out + 3072 + 262144;
    float* T1o = out + 3072 + 262144 + 524288;

    // ---- 1: unfolds -----------------------------------------------------
    unfold3_kernel<<<1024, 256, 0, stream>>>(lrsr_lv3, 256, 32, lr3n, nullptr, 2);
    unfold3_kernel<<<1024, 256, 0, stream>>>(refsr_lv3, 256, 32, A3, inv3, 1);
    unfold3_kernel<<<4096, 256, 0, stream>>>(refsr_lv2, 128, 64, u2, nullptr, 0);
    unfold3_kernel<<<16384, 256, 0, stream>>>(refsr_lv1, 64, 128, u1, nullptr, 0);

    // ---- 2: weight transposes to swizzled f16 ----------------------------
    pack_w_swizzled<<<(2304 * 1152 + 255) / 256, 256, 0, stream>>>(W1, B1t, 2304, 1152);
    pack_w_swizzled<<<(2304 * 576 + 255) / 256, 256, 0, stream>>>(W2, B2t, 2304, 576);

    // ---- 3: feature GEMMs (WMMA), f16 out, fused bias --------------------
    gemm_f16_wmma<<<dim3(18, 32), 256, 0, stream>>>(u2, B1t, b1, nullptr, r2, 4096, 2304, 1152);
    gemm_f16_wmma<<<dim3(18, 128), 256, 0, stream>>>(u1, B2t, b2, nullptr, r1, 16384, 2304, 576);

    // ---- 4: row inverse norms -------------------------------------------
    row_invnorm_f16<<<4096, 256, 0, stream>>>(r2, 2304, inv2);
    row_invnorm_f16<<<16384, 256, 0, stream>>>(r1, 2304, inv1);

    // ---- 5: score GEMMs + column max/argmax ------------------------------
    gemm_f16_wmma<<<dim3(8, 8), 256, 0, stream>>>(A3, lr3n, nullptr, scores, nullptr, 1024, 1024, 2304);
    colmax_kernel<<<1024, 256, 0, stream>>>(scores, inv3, 1024, 1024, S3o, arg);

    gemm_f16_wmma<<<dim3(8, 32), 256, 0, stream>>>(r2, lr3n, nullptr, scores, nullptr, 4096, 1024, 2304);
    colmax_kernel<<<1024, 256, 0, stream>>>(scores, inv2, 4096, 1024, S2o, nullptr);

    gemm_f16_wmma<<<dim3(8, 128), 256, 0, stream>>>(r1, lr3n, nullptr, scores, nullptr, 16384, 1024, 2304);
    colmax_kernel<<<1024, 256, 0, stream>>>(scores, inv1, 16384, 1024, S1o, nullptr);

    // ---- 6: gather + fold transfers -------------------------------------
    gather_fold_kernel<<<(256 * 32 * 32 + 255) / 256, 256, 0, stream>>>(
        ref_lv3, arg, 256, 32, 3, 1, 1, T3o);
    gather_fold_kernel<<<(128 * 64 * 64 + 255) / 256, 256, 0, stream>>>(
        ref_lv2, arg, 128, 64, 6, 2, 2, T2o);
    gather_fold_kernel<<<(64 * 128 * 128 + 255) / 256, 256, 0, stream>>>(
        ref_lv1, arg, 64, 128, 12, 4, 4, T1o);
}
